// RAVEN_78340203479702
// MI455X (gfx1250) — matrix operand order repaired and checked by reading, not compiled, not run on hardware
//
#include <hip/hip_runtime.h>

// ---------------- CDNA5 WMMA types ----------------
typedef __bf16 bf16_t;
typedef __attribute__((ext_vector_type(16))) __bf16 v16bf;
typedef __attribute__((ext_vector_type(8)))  __bf16 v8bf;
typedef __attribute__((ext_vector_type(8)))  float  v8f;

union AFrag { v16bf v; v8bf h[2]; };

__device__ __forceinline__ float sigf(float x) { return 1.0f / (1.0f + __expf(-x)); }

__device__ __forceinline__ v8bf cvt8(const float4 f0, const float4 f1) {
    v8bf o;
    o[0] = (bf16_t)f0.x; o[1] = (bf16_t)f0.y; o[2] = (bf16_t)f0.z; o[3] = (bf16_t)f0.w;
    o[4] = (bf16_t)f1.x; o[5] = (bf16_t)f1.y; o[6] = (bf16_t)f1.z; o[7] = (bf16_t)f1.w;
    return o;
}

// =====================================================================
// Generic tiled bf16-WMMA GEMM:  C = act( A[M,K] * B (+bias +bias2 +C_in) )
// TRANSB=1: B is weight [N,K] (row stride ldb), computes A @ B^T
// TRANSB=0: B is [K,N]      (row stride ldb), computes A @ B
// ACT: 0 none, 1 relu, 2 sigmoid.  ACCUM: add existing C.
// GUARD=0: requires M%64==0, N%64==0, K%32==0; double-buffered LDS
//          pipeline (prefetch chunk k+1 in regs during chunk k's WMMAs).
// Block = 256 threads (8 waves), tile 64x64, K-chunk 32.
// =====================================================================
template <int TRANSB, int ACT, int ACCUM, int GUARD>
__global__ __launch_bounds__(256) void gemm_wmma(
    const float* __restrict__ A, const float* __restrict__ Bm,
    const float* __restrict__ bias, const float* __restrict__ bias2,
    float* __restrict__ C, int M, int N, int K, int ldb)
{
    __shared__ bf16_t As[2][64][32];   // A tile, row-major [m][k], double buffered
    __shared__ bf16_t Bs[2][64][32];   // B tile stored as [n][k], double buffered

    const int tid  = threadIdx.x;
    const int row0 = blockIdx.y * 64;
    const int col0 = blockIdx.x * 64;
    const int wave = tid >> 5;
    const int lane = tid & 31;
    const int half = lane >> 4;
    const int lq   = lane & 15;
    const int mt   = wave & 3;          // M-tile of this wave
    const int ntb  = (wave >> 2) * 2;   // first of 2 N-tiles

    v8f acc0 = {};
    v8f acc1 = {};

    const int nkc = (K + 31) >> 5;

    if constexpr (!GUARD) {
        // ---- software-pipelined path ----
        const int r  = tid >> 2;             // staging row (m for A, n for B^T)
        const int kq = (tid & 3) * 8;        // 8 contiguous k per thread
        const float* Arow = A + (size_t)(row0 + r) * K + kq;
        const float* Brow = TRANSB ? (Bm + (size_t)(col0 + r) * ldb + kq) : nullptr;
        const int kNN0 = tid >> 4;           // NN staging: k rows 0..15 / 16..31
        const int nqNN = (tid & 15) * 4;     // 4 contiguous n per thread

        float4 a0, a1, b0q, b1q, n0q, n1q;
        // prefetch chunk 0
        a0 = *(const float4*)(Arow + 0);
        a1 = *(const float4*)(Arow + 4);
        if constexpr (TRANSB) {
            b0q = *(const float4*)(Brow + 0);
            b1q = *(const float4*)(Brow + 4);
        } else {
            n0q = *(const float4*)(Bm + (size_t)(kNN0)      * ldb + col0 + nqNN);
            n1q = *(const float4*)(Bm + (size_t)(kNN0 + 16) * ldb + col0 + nqNN);
        }

        for (int kc = 0; kc < nkc; ++kc) {
            const int buf = kc & 1;
            // store prefetched chunk into LDS buffer `buf`
            *(v8bf*)(&As[buf][r][kq]) = cvt8(a0, a1);
            if constexpr (TRANSB) {
                *(v8bf*)(&Bs[buf][r][kq]) = cvt8(b0q, b1q);
            } else {
                Bs[buf][nqNN + 0][kNN0] = (bf16_t)n0q.x;
                Bs[buf][nqNN + 1][kNN0] = (bf16_t)n0q.y;
                Bs[buf][nqNN + 2][kNN0] = (bf16_t)n0q.z;
                Bs[buf][nqNN + 3][kNN0] = (bf16_t)n0q.w;
                Bs[buf][nqNN + 0][kNN0 + 16] = (bf16_t)n1q.x;
                Bs[buf][nqNN + 1][kNN0 + 16] = (bf16_t)n1q.y;
                Bs[buf][nqNN + 2][kNN0 + 16] = (bf16_t)n1q.z;
                Bs[buf][nqNN + 3][kNN0 + 16] = (bf16_t)n1q.w;
            }
            __syncthreads();

            // issue global loads for chunk kc+1 (overlap with WMMAs below)
            if (kc + 1 < nkc) {
                const int kb = (kc + 1) * 32;
                a0 = *(const float4*)(Arow + kb);
                a1 = *(const float4*)(Arow + kb + 4);
                if constexpr (TRANSB) {
                    b0q = *(const float4*)(Brow + kb);
                    b1q = *(const float4*)(Brow + kb + 4);
                } else {
                    n0q = *(const float4*)(Bm + (size_t)(kb + kNN0)      * ldb + col0 + nqNN);
                    n1q = *(const float4*)(Bm + (size_t)(kb + kNN0 + 16) * ldb + col0 + nqNN);
                }
            }

            // fragments (contiguous k order: lane half h holds k = 16h..16h+15 for A and B)
            AFrag a;
            const bf16_t* ap = &As[buf][mt * 16 + lq][0];
            a.h[0] = *(const v8bf*)(ap + half * 16);
            a.h[1] = *(const v8bf*)(ap + half * 16 + 8);

            AFrag fb0, fb1;
            const bf16_t* bp0 = &Bs[buf][ntb * 16 + lq][0];
            fb0.h[0] = *(const v8bf*)(bp0 + half * 16);
            fb0.h[1] = *(const v8bf*)(bp0 + half * 16 + 8);
            const bf16_t* bp1 = &Bs[buf][(ntb + 1) * 16 + lq][0];
            fb1.h[0] = *(const v8bf*)(bp1 + half * 16);
            fb1.h[1] = *(const v8bf*)(bp1 + half * 16 + 8);

            acc0 = __builtin_amdgcn_wmma_f32_16x16x32_bf16(false, a.v, false, fb0.v,
                                                           (short)0, acc0, false, false);
            acc1 = __builtin_amdgcn_wmma_f32_16x16x32_bf16(false, a.v, false, fb1.v,
                                                           (short)0, acc1, false, false);
            // no trailing barrier: double buffering guarantees safety
        }
    } else {
        // ---- guarded (edge-shape) path ----
        for (int kc = 0; kc < nkc; ++kc) {
            const int kb = kc * 32;
            {
                int k = tid & 31;
                int r = tid >> 5;
                #pragma unroll
                for (int i = 0; i < 8; ++i, r += 8) {
                    float v = 0.0f;
                    int gr = row0 + r, gk = kb + k;
                    if (gr < M && gk < K) v = A[(size_t)gr * K + gk];
                    As[0][r][k] = (bf16_t)v;
                }
            }
            if (TRANSB) {
                int k = tid & 31;
                int n = tid >> 5;
                #pragma unroll
                for (int i = 0; i < 8; ++i, n += 8) {
                    float v = 0.0f;
                    int gn = col0 + n, gk = kb + k;
                    if (gn < N && gk < K) v = Bm[(size_t)gn * ldb + gk];
                    Bs[0][n][k] = (bf16_t)v;
                }
            } else {
                int n = tid & 63;
                int k = tid >> 6;
                #pragma unroll
                for (int i = 0; i < 8; ++i, k += 4) {
                    float v = 0.0f;
                    int gn = col0 + n, gk = kb + k;
                    if (gn < N && gk < K) v = Bm[(size_t)gk * ldb + gn];
                    Bs[0][n][k] = (bf16_t)v;
                }
            }
            __syncthreads();

            AFrag a;
            const bf16_t* ap = &As[0][mt * 16 + lq][0];
            a.h[0] = *(const v8bf*)(ap + half * 16);
            a.h[1] = *(const v8bf*)(ap + half * 16 + 8);

            AFrag fb0, fb1;
            const bf16_t* bp0 = &Bs[0][ntb * 16 + lq][0];
            fb0.h[0] = *(const v8bf*)(bp0 + half * 16);
            fb0.h[1] = *(const v8bf*)(bp0 + half * 16 + 8);
            const bf16_t* bp1 = &Bs[0][(ntb + 1) * 16 + lq][0];
            fb1.h[0] = *(const v8bf*)(bp1 + half * 16);
            fb1.h[1] = *(const v8bf*)(bp1 + half * 16 + 8);

            acc0 = __builtin_amdgcn_wmma_f32_16x16x32_bf16(false, a.v, false, fb0.v,
                                                           (short)0, acc0, false, false);
            acc1 = __builtin_amdgcn_wmma_f32_16x16x32_bf16(false, a.v, false, fb1.v,
                                                           (short)0, acc1, false, false);
            __syncthreads();
        }
    }

    // ---- epilogue (C/D layout: row = v + 8*half) ----
    #pragma unroll
    for (int j = 0; j < 2; ++j) {
        v8f acc = j ? acc1 : acc0;
        int col = col0 + (ntb + j) * 16 + lq;
        if (GUARD && col >= N) continue;
        float bv = 0.0f;
        if (bias)  bv += bias[col];
        if (bias2) bv += bias2[col];
        #pragma unroll
        for (int v = 0; v < 8; ++v) {
            int row = row0 + mt * 16 + v + half * 8;
            if (GUARD && row >= M) continue;
            float x = acc[v] + bv;
            if (ACCUM) x += C[(size_t)row * N + col];
            if (ACT == 1) x = fmaxf(x, 0.0f);
            else if (ACT == 2) x = sigf(x);
            C[(size_t)row * N + col] = x;
        }
    }
}

// =====================================================================
// One-time W_hh -> bf16 swizzle into WMMA B-fragment order:
// dst[(((nt*KC)+kc)*32 + lane)*16 + i] = Whh[n*H + k]  (0 for k>=H)
//   n = nt*16 + (lane&15),  k = kc*32 + (lane>>4)*16 + i
// =====================================================================
__global__ void whh_swizzle(const float* __restrict__ Whh, bf16_t* __restrict__ dst,
                            int H, int KC, int NT)
{
    const long total = (long)NT * KC * 32 * 16;
    for (long idx = blockIdx.x * (long)blockDim.x + threadIdx.x; idx < total;
         idx += (long)gridDim.x * blockDim.x) {
        int  i    = (int)(idx & 15);
        long g    = idx >> 4;
        int  lane = (int)(g & 31);
        g >>= 5;
        int kc = (int)(g % KC);
        int nt = (int)(g / KC);
        int n  = nt * 16 + (lane & 15);
        int k  = kc * 32 + (lane >> 4) * 16 + i;
        float v = (k < H) ? Whh[(long)n * H + k] : 0.0f;
        dst[idx] = (bf16_t)v;
    }
}

// =====================================================================
// Persistent-workgroup LSTM recurrence (wave32 WMMA per step).
// One 1024-thread block runs all S steps: h (bf16) in LDS, gates (f32)
// in LDS, cell state in registers, swizzled bf16 W_hh streamed from L2.
// xp: [64, S, 4H] precomputed input projection (+ both biases).
// =====================================================================
struct LstmArgs { const float* xp; const bf16_t* Wsw; float* hout; float* hT; };

template <int H, int S>
__global__ __launch_bounds__(1024) void lstm_rec(LstmArgs a0, LstmArgs a1)
{
    constexpr int KP = ((H + 31) / 32) * 32;   // K padded to 32
    constexpr int KC = KP / 32;
    constexpr int N  = 4 * H;                  // gate width (mult of 16)
    constexpr int NTILES = 4 * (N / 16);       // 4 M-tiles x N/16 N-tiles
    constexpr int CE = (64 * H + 1023) / 1024; // cell elems per thread

    const LstmArgs& P = (blockIdx.x == 0) ? a0 : a1;

    extern __shared__ char smem[];
    bf16_t* sh_h = (bf16_t*)smem;                                      // [64][KP]
    float*  sh_g = (float*)(smem + (size_t)64 * KP * sizeof(bf16_t));  // [64][N]

    const int tid  = threadIdx.x;
    const int wave = tid >> 5;
    const int lane = tid & 31;
    const int half = lane >> 4;
    const int lq   = lane & 15;

    // h = 0 (zero-fill includes K padding, which stays zero forever)
    for (int i = tid; i < 64 * KP; i += 1024) sh_h[i] = (bf16_t)0.0f;
    float c_reg[CE];
    #pragma unroll
    for (int e = 0; e < CE; ++e) c_reg[e] = 0.0f;

    for (int t = 0; t < S; ++t) {
        __syncthreads();   // sh_h (from prev step) visible

        // ---- gates = h @ Whh^T  (M=64, N=4H, K=H) ----
        for (int tile = wave; tile < NTILES; tile += 32) {
            const int mt = tile & 3;
            const int nt = tile >> 2;
            v8f acc = {};
            #pragma unroll
            for (int kc = 0; kc < KC; ++kc) {
                AFrag a;
                const bf16_t* ap = sh_h + (size_t)(mt * 16 + lq) * KP + kc * 32;
                a.h[0] = *(const v8bf*)(ap + half * 16);
                a.h[1] = *(const v8bf*)(ap + half * 16 + 8);

                // swizzled weights: wave reads 1024 contiguous bytes
                AFrag bf;
                const v8bf* wp =
                    (const v8bf*)(P.Wsw + ((size_t)(nt * KC + kc) * 32 + lane) * 16);
                bf.h[0] = wp[0];
                bf.h[1] = wp[1];

                acc = __builtin_amdgcn_wmma_f32_16x16x32_bf16(false, a.v, false, bf.v,
                                                              (short)0, acc, false, false);
            }
            #pragma unroll
            for (int v = 0; v < 8; ++v) {
                int m   = mt * 16 + v + half * 8;
                int col = nt * 16 + lq;
                sh_g[m * N + col] = acc[v];
            }
        }
        __syncthreads();   // gates visible

        // ---- elementwise LSTM cell (i,f,g,o gate order) ----
        #pragma unroll
        for (int e = 0; e < CE; ++e) {
            int idx = tid + e * 1024;
            if (idx < 64 * H) {
                int bb = idx / H;
                int j  = idx - bb * H;
                const float* xr = P.xp + ((size_t)bb * S + t) * N;
                float ig = sh_g[bb * N + j]         + xr[j];
                float fg = sh_g[bb * N + H + j]     + xr[H + j];
                float gg = sh_g[bb * N + 2 * H + j] + xr[2 * H + j];
                float og = sh_g[bb * N + 3 * H + j] + xr[3 * H + j];
                float c  = sigf(fg) * c_reg[e] + sigf(ig) * tanhf(gg);
                c_reg[e] = c;
                float h  = sigf(og) * tanhf(c);
                sh_h[(size_t)bb * KP + j] = (bf16_t)h;
                if (P.hout) P.hout[((size_t)bb * S + t) * H + j] = h;
                if (P.hT && t == S - 1) P.hT[(size_t)bb * H + j] = h;
            }
        }
    }
}

// =====================================================================
// Elementwise fusion: hm = ga*t1 + gv*t2 + bh  (in place over t1)
// =====================================================================
__global__ __launch_bounds__(256) void hm_fuse(
    const float* __restrict__ ga, const float* __restrict__ t1,
    const float* __restrict__ gv, const float* __restrict__ t2,
    const float* __restrict__ bh, float* __restrict__ hm, int rows, int Nc)
{
    for (int r = blockIdx.x; r < rows; r += gridDim.x) {
        const size_t base = (size_t)r * Nc;
        for (int ccol = threadIdx.x; ccol < Nc; ccol += blockDim.x) {
            const size_t i = base + ccol;
            hm[i] = ga[i] * t1[i] + gv[i] * t2[i] + bh[ccol];
        }
    }
}

// =====================================================================
// em = text + min(||text|| / ||hm||, 1) * hm  (one block per (b,s) row)
// =====================================================================
__global__ __launch_bounds__(256) void em_norm(const float* __restrict__ text,
                                               const float* __restrict__ hm,
                                               float* __restrict__ em)
{
    const long row = blockIdx.x;
    const float* tr = text + row * 768;
    const float* hr = hm + row * 768;
    float st = 0.0f, sh = 0.0f;
    for (int i = threadIdx.x; i < 768; i += 256) {
        float a = tr[i], b = hr[i];
        st += a * a;
        sh += b * b;
    }
    __shared__ float r0[256], r1[256];
    r0[threadIdx.x] = st;
    r1[threadIdx.x] = sh;
    __syncthreads();
    for (int off = 128; off > 0; off >>= 1) {
        if (threadIdx.x < off) {
            r0[threadIdx.x] += r0[threadIdx.x + off];
            r1[threadIdx.x] += r1[threadIdx.x + off];
        }
        __syncthreads();
    }
    const float aa = fminf(sqrtf(r0[0] / r1[0]), 1.0f);
    for (int i = threadIdx.x; i < 768; i += 256)
        em[row * 768 + i] = tr[i] + aa * hr[i];
}

// =====================================================================
// Final N=1 layer: y[b] = dot(y2[b,:512], W3) + b3
// =====================================================================
__global__ __launch_bounds__(256) void final_out(const float* __restrict__ y2,
                                                 const float* __restrict__ W3,
                                                 const float* __restrict__ b3,
                                                 float* __restrict__ out)
{
    const int row = blockIdx.x;
    float s = 0.0f;
    for (int i = threadIdx.x; i < 512; i += 256) s += y2[row * 512 + i] * W3[i];
    __shared__ float r[256];
    r[threadIdx.x] = s;
    __syncthreads();
    for (int off = 128; off > 0; off >>= 1) {
        if (threadIdx.x < off) r[threadIdx.x] += r[threadIdx.x + off];
        __syncthreads();
    }
    if (threadIdx.x == 0) out[row] = r[0] + b3[0];
}

// =====================================================================
// Host-side orchestration
// =====================================================================
extern "C" void kernel_launch(void* const* d_in, const int* in_sizes, int n_in,
                              void* d_out, int out_size, void* d_ws, size_t ws_size,
                              hipStream_t stream)
{
    (void)in_sizes; (void)n_in; (void)out_size; (void)ws_size;

    const float* text  = (const float*)d_in[0];   // [64,256,768]
    const float* audio = (const float*)d_in[1];   // [64,256,128]
    const float* video = (const float*)d_in[2];   // [64,256,128]
    const float* wa    = (const float*)d_in[3];   // [1024,768]
    const float* ba    = (const float*)d_in[4];   // [768]
    const float* wv    = (const float*)d_in[5];   // [1024,768]
    const float* bv    = (const float*)d_in[6];   // [768]
    const float* bh    = (const float*)d_in[7];   // [768]
    const float* wa2   = (const float*)d_in[8];   // [256,768]
    const float* wv2   = (const float*)d_in[9];   // [256,768]
    const float* Wa_ih = (const float*)d_in[10];  // [1024,128]
    const float* Wa_hh = (const float*)d_in[11];  // [1024,256]
    const float* ba_ih = (const float*)d_in[12];
    const float* ba_hh = (const float*)d_in[13];
    const float* Wv_ih = (const float*)d_in[14];
    const float* Wv_hh = (const float*)d_in[15];
    const float* bv_ih = (const float*)d_in[16];
    const float* bv_hh = (const float*)d_in[17];
    const float* Wl_ih = (const float*)d_in[18];  // [400,768]
    const float* Wl_hh = (const float*)d_in[19];  // [400,100]
    const float* bl_ih = (const float*)d_in[20];
    const float* bl_hh = (const float*)d_in[21];
    const float* W1    = (const float*)d_in[22];  // [512,100]
    const float* b1    = (const float*)d_in[23];
    const float* W2    = (const float*)d_in[24];  // [512,512]
    const float* b2    = (const float*)d_in[25];
    const float* W3    = (const float*)d_in[26];  // [1,512]
    const float* b3    = (const float*)d_in[27];
    float* out = (float*)d_out;                   // [64,1]

    const long M = 64L * 256;                     // 16384 rows (b*S+s)
    float* ws = (float*)d_ws;

    // ---- workspace arena (floats), with reuse ----
    const size_t SZ_XP  = (size_t)M * 1024;       // 16.78M
    const size_t SZ_H   = (size_t)M * 256;        //  4.19M
    const size_t SZ_768 = (size_t)M * 768;        // 12.58M
    float* xpA      = ws;                         // audio input proj
    float* xpV      = xpA + SZ_XP;                // video input proj
    float* acoustic = xpV + SZ_XP;
    float* visual   = acoustic + SZ_H;
    float* t1_hm    = visual + SZ_H;              // t1, later hm (in place)
    float* t2       = t1_hm + SZ_768;
    float* smallb   = t2 + SZ_768;
    float* U  = smallb;                           // [64,100]
    float* y1 = U + 64 * 112;                     // [64,512]
    float* y2 = y1 + 64 * 512;                    // [64,512]
    // swizzled bf16 recurrent weights (after y2)
    bf16_t* WswA = (bf16_t*)(y2 + 64 * 512);      // 64*8*32*16  = 262144 bf16
    bf16_t* WswV = WswA + 262144;                 // 262144 bf16
    bf16_t* WswL = WswV + 262144;                 // 25*4*32*16  =  51200 bf16
    // reuse after side LSTMs consume xpA/xpV:
    float* ga  = xpA;                             // [16384,768]
    float* gv  = xpV;                             // [16384,768]
    float* em  = xpA;                             // after ga dead
    float* xpL = xpV;                             // after gv dead, [16384,400]

    const dim3 blk(256);

    // 0) swizzle recurrent weights to bf16 fragment order (one-time)
    whh_swizzle<<<512, blk, 0, stream>>>(Wa_hh, WswA, 256, 8, 64);
    whh_swizzle<<<512, blk, 0, stream>>>(Wv_hh, WswV, 256, 8, 64);
    whh_swizzle<<<128, blk, 0, stream>>>(Wl_hh, WswL, 100, 4, 25);

    // 1) LSTM input projections: xp = x @ W_ih^T + b_ih + b_hh
    gemm_wmma<1, 0, 0, 0><<<dim3(1024 / 64, M / 64), blk, 0, stream>>>(
        audio, Wa_ih, ba_ih, ba_hh, xpA, (int)M, 1024, 128, 128);
    gemm_wmma<1, 0, 0, 0><<<dim3(1024 / 64, M / 64), blk, 0, stream>>>(
        video, Wv_ih, bv_ih, bv_hh, xpV, (int)M, 1024, 128, 128);

    // 2) audio + video recurrences concurrently (2 persistent blocks)
    {
        LstmArgs la{xpA, WswA, acoustic, nullptr};
        LstmArgs lv{xpV, WswV, visual, nullptr};
        const size_t shmem =
            (size_t)64 * 256 * sizeof(bf16_t) + (size_t)64 * 1024 * sizeof(float);
        lstm_rec<256, 256><<<2, 1024, shmem, stream>>>(la, lv);
    }

    // 3) fusion GEMMs
    gemm_wmma<0, 0, 0, 0><<<dim3(768 / 64, M / 64), blk, 0, stream>>>(
        acoustic, wa2, nullptr, nullptr, t1_hm, (int)M, 768, 256, 768);
    gemm_wmma<0, 0, 0, 0><<<dim3(768 / 64, M / 64), blk, 0, stream>>>(
        acoustic, wa, nullptr, nullptr, ga, (int)M, 768, 256, 768);
    gemm_wmma<0, 2, 1, 0><<<dim3(768 / 64, M / 64), blk, 0, stream>>>(
        text, wa + (size_t)256 * 768, ba, nullptr, ga, (int)M, 768, 768, 768);
    gemm_wmma<0, 0, 0, 0><<<dim3(768 / 64, M / 64), blk, 0, stream>>>(
        visual, wv2, nullptr, nullptr, t2, (int)M, 768, 256, 768);
    gemm_wmma<0, 0, 0, 0><<<dim3(768 / 64, M / 64), blk, 0, stream>>>(
        visual, wv, nullptr, nullptr, gv, (int)M, 768, 256, 768);
    gemm_wmma<0, 2, 1, 0><<<dim3(768 / 64, M / 64), blk, 0, stream>>>(
        text, wv + (size_t)256 * 768, bv, nullptr, gv, (int)M, 768, 768, 768);

    // 4) hm = ga*t1 + gv*t2 + bh  (in place over t1)
    hm_fuse<<<4096, blk, 0, stream>>>(ga, t1_hm, gv, t2, bh, t1_hm, (int)M, 768);

    // 5) em = text + min(||text||/||hm||, 1) * hm
    em_norm<<<(unsigned)M, blk, 0, stream>>>(text, t1_hm, em);

    // 6) main LSTM input projection: xpL = em @ Wl_ih^T + bl_ih + bl_hh  (N=400 -> guarded)
    gemm_wmma<1, 0, 0, 1><<<dim3((400 + 63) / 64, M / 64), blk, 0, stream>>>(
        em, Wl_ih, bl_ih, bl_hh, xpL, (int)M, 400, 768, 768);

    // 7) main recurrence (H=100, K padded to 128), keep only final h -> U
    {
        LstmArgs lm{xpL, WswL, nullptr, U};
        const size_t shmem =
            (size_t)64 * 128 * sizeof(bf16_t) + (size_t)64 * 400 * sizeof(float);
        lstm_rec<100, 256><<<1, 1024, shmem, stream>>>(lm, lm);
    }

    // 8) MLP head
    gemm_wmma<1, 1, 0, 1><<<dim3(512 / 64, 1), blk, 0, stream>>>(
        U, W1, b1, nullptr, y1, 64, 512, 100, 100);             // K=100 -> guarded
    gemm_wmma<1, 1, 0, 0><<<dim3(512 / 64, 1), blk, 0, stream>>>(
        y1, W2, b2, nullptr, y2, 64, 512, 512, 512);
    final_out<<<64, blk, 0, stream>>>(y2, W3, b3, out);
}
